// EEGMamba_79894981640340
// MI455X (gfx1250) — compile-verified, matrix operand order checked by reading, not run on hardware
//
#include <hip/hip_runtime.h>
#include <hip/hip_bf16.h>
#include <stdint.h>

#define DEV __device__ __forceinline__

typedef __attribute__((ext_vector_type(16))) __bf16        v16bf;
typedef __attribute__((ext_vector_type(8)))  float         v8f;
typedef __attribute__((ext_vector_type(8)))  unsigned short u16x8;

// ---- model dims ----
#define B_     8
#define C_     19
#define L_     3000
#define D_     512
#define DI_    1024
#define DS_    16
#define DTR_   32
#define KC_    4
#define PATCH_ 20
#define NB_    4
#define E_     8
#define F_     2048
#define NC_    5
#define LC_    151      // cls + 150 patch tokens
#define LT_    150
#define MV_    1208     // valid rows = B_*LC_
#define MP_    1216     // padded rows (multiple of 64), also covers B_*LT_=1200

#define LSTRIDE 40      // LDS row stride in bf16 elems (80B) -> conflict-free frag reads

DEV unsigned short f2bf(float f) {
  unsigned u = __builtin_bit_cast(unsigned, f);
  u += 0x7FFFu + ((u >> 16) & 1u);          // round-to-nearest-even
  return (unsigned short)(u >> 16);
}
DEV float bf2f(unsigned short h) {
  return __builtin_bit_cast(float, (unsigned)h << 16);
}
DEV float silu_(float v) { return v / (1.f + __expf(-v)); }

// ---------------- elementwise / stem kernels ----------------

__global__ void k_spatial(const float* __restrict__ x, const float* __restrict__ w,
                          const float* __restrict__ bias, float* __restrict__ y) {
  long i = (long)blockIdx.x * blockDim.x + threadIdx.x;
  long n = (long)B_ * D_ * L_;
  if (i >= n) return;
  int l = (int)(i % L_);
  int d = (int)((i / L_) % D_);
  int b = (int)(i / ((long)D_ * L_));
  float acc = bias[d];
  const float* xp = x + (long)b * C_ * L_ + l;
  const float* wp = w + d * C_;
#pragma unroll
  for (int c = 0; c < C_; ++c) acc += wp[c] * xp[(long)c * L_];
  y[i] = silu_(acc);
}

__global__ void k_im2col(const float* __restrict__ y, unsigned short* __restrict__ p) {
  const long KK = (long)D_ * PATCH_;
  long i = (long)blockIdx.x * blockDim.x + threadIdx.x;
  long n = (long)MP_ * KK;
  if (i >= n) return;
  int r = (int)(i / KK);
  int k = (int)(i % KK);
  if (r >= B_ * LT_) { p[i] = 0; return; }
  int b = r / LT_, t = r % LT_;
  int di = k / PATCH_, pp = k % PATCH_;
  p[i] = f2bf(y[(long)b * D_ * L_ + (long)di * L_ + t * PATCH_ + pp]);
}

__global__ void k_f2bf(const float* __restrict__ s, unsigned short* __restrict__ d, long n) {
  long i = (long)blockIdx.x * blockDim.x + threadIdx.x;
  long stride = (long)gridDim.x * blockDim.x;
  for (; i < n; i += stride) d[i] = f2bf(s[i]);
}

__global__ void k_build_T(const float* __restrict__ tok, const float* __restrict__ cls,
                          float* __restrict__ T) {
  long i = (long)blockIdx.x * blockDim.x + threadIdx.x;
  long n = (long)MP_ * D_;
  if (i >= n) return;
  int r = (int)(i / D_), d = (int)(i % D_);
  float v = 0.f;
  if (r < MV_) {
    int b = r / LC_, t = r % LC_;
    v = (t == 0) ? cls[d] : tok[((long)(b * LT_ + t - 1)) * D_ + d];
  }
  T[i] = v;
}

__global__ void k_rms_bf16(const float* __restrict__ T, const float* __restrict__ w,
                           unsigned short* __restrict__ o) {
  __shared__ float red[256];
  int r = blockIdx.x;
  const float* tp = T + (long)r * D_;
  float x0 = tp[threadIdx.x], x1 = tp[threadIdx.x + 256];
  red[threadIdx.x] = x0 * x0 + x1 * x1;
  __syncthreads();
  for (int s = 128; s > 0; s >>= 1) {
    if (threadIdx.x < s) red[threadIdx.x] += red[threadIdx.x + s];
    __syncthreads();
  }
  float rinv = rsqrtf(red[0] / (float)D_ + 1e-6f);
  o[(long)r * D_ + threadIdx.x]       = f2bf(x0 * w[threadIdx.x] * rinv);
  o[(long)r * D_ + threadIdx.x + 256] = f2bf(x1 * w[threadIdx.x + 256] * rinv);
}

__global__ void k_dwconv_silu(const float* __restrict__ xz, const float* __restrict__ cw,
                              const float* __restrict__ cb, float* __restrict__ xc, int flip) {
  long i = (long)blockIdx.x * blockDim.x + threadIdx.x;
  long n = (long)MP_ * DI_;
  if (i >= n) return;
  int r = (int)(i / DI_), di = (int)(i % DI_);
  if (r >= MV_) { xc[i] = 0.f; return; }
  int b = r / LC_, t = r % LC_;
  float acc = cb[di];
#pragma unroll
  for (int j = 0; j < KC_; ++j) {
    int tt = t - (KC_ - 1) + j;
    if (tt >= 0) {
      int ts = flip ? (LC_ - 1 - tt) : tt;
      acc += cw[di * KC_ + j] * xz[((long)(b * LC_ + ts)) * (2 * DI_) + di];
    }
  }
  xc[i] = silu_(acc);
}

__global__ void k_extract_dtr(const float* __restrict__ proj, unsigned short* __restrict__ d) {
  long i = (long)blockIdx.x * blockDim.x + threadIdx.x;
  long n = (long)MP_ * DTR_;
  if (i >= n) return;
  int r = (int)(i / DTR_), k = (int)(i % DTR_);
  d[i] = f2bf(proj[(long)r * 64 + k]);
}

__global__ void k_ssm(const float* __restrict__ xc, const float* __restrict__ dt,
                      const float* __restrict__ proj, const float* __restrict__ Alog,
                      const float* __restrict__ Dp, float* __restrict__ ys) {
  int tid = blockIdx.x * blockDim.x + threadIdx.x;
  if (tid >= B_ * DI_) return;
  int b = tid / DI_, di = tid % DI_;
  float A[DS_], h[DS_];
#pragma unroll
  for (int s = 0; s < DS_; ++s) { A[s] = -__expf(Alog[di * DS_ + s]); h[s] = 0.f; }
  float dp = Dp[di];
  for (int t = 0; t < LC_; ++t) {
    long r = (long)b * LC_ + t;
    float dtv = dt[r * DI_ + di];
    float xv  = xc[r * DI_ + di];
    float dx  = dtv * xv;
    const float* pr = proj + r * 64;
    float yv = 0.f;
#pragma unroll
    for (int s = 0; s < DS_; ++s) {
      h[s] = __expf(dtv * A[s]) * h[s] + dx * pr[32 + s];
      yv  += h[s] * pr[48 + s];
    }
    ys[r * DI_ + di] = yv + dp * xv;
  }
}

__global__ void k_combine(const float* __restrict__ ysf, const float* __restrict__ ysb,
                          const float* __restrict__ xz, unsigned short* __restrict__ ub) {
  long i = (long)blockIdx.x * blockDim.x + threadIdx.x;
  long n = (long)MP_ * DI_;
  if (i >= n) return;
  int r = (int)(i / DI_), di = (int)(i % DI_);
  if (r >= MV_) { ub[i] = 0; return; }
  int b = r / LC_, t = r % LC_;
  float zf = xz[(long)r * (2 * DI_) + DI_ + di];
  float v = (ysf[i] + ysb[((long)(b * LC_ + (LC_ - 1 - t))) * DI_ + di]) * silu_(zf);
  ub[i] = f2bf(v);
}

__global__ void k_gate(const unsigned short* __restrict__ hb, const float* __restrict__ gw,
                       float* __restrict__ gates) {
  int r = blockIdx.x * blockDim.x + threadIdx.x;
  if (r >= MP_) return;
  if (r >= MV_) { for (int e = 0; e < E_; ++e) gates[(long)r * E_ + e] = 0.f; return; }
  float g[E_];
  const unsigned short* hp = hb + (long)r * D_;
  for (int e = 0; e < E_; ++e) {
    float acc = 0.f;
    const float* wp = gw + e * D_;
    for (int d = 0; d < D_; ++d) acc += bf2f(hp[d]) * wp[d];
    g[e] = acc;
  }
  int bi = 0; float bv = g[0];
  for (int e = 1; e < E_; ++e) if (g[e] > bv) { bv = g[e]; bi = e; }
  int si = -1; float sv = -1e30f;
  for (int e = 0; e < E_; ++e) if (e != bi && g[e] > sv) { sv = g[e]; si = e; }
  float p2 = __expf(sv - bv);
  float inv = 1.f / (1.f + p2);
  for (int e = 0; e < E_; ++e) gates[(long)r * E_ + e] = 0.f;
  gates[(long)r * E_ + bi] = inv;
  gates[(long)r * E_ + si] = p2 * inv;
}

__global__ void k_cls(const float* __restrict__ T, const float* __restrict__ cw,
                      const float* __restrict__ cb, float* __restrict__ out) {
  int tid = threadIdx.x;
  if (tid >= B_ * NC_) return;
  int b = tid / NC_, c = tid % NC_;
  const float* tp = T + (long)(b * LC_) * D_;
  const float* wp = cw + c * D_;
  float acc = cb[c];
  for (int d = 0; d < D_; ++d) acc += tp[d] * wp[d];
  out[b * NC_ + c] = acc;
}

// ---------------- WMMA bf16 GEMM (LDS double-buffered B, pipelined A) ----------------
// C[M,N] = op(A[M,K] * B[N,K]^T + bias) * rowScale  (optionally += into C)
// A,B bf16 row-major (K contiguous); M,N multiples of 64; K multiple of 32.
// Block = 128 threads (4 waves); block tile 64x64; wave tile 16x64.
// B tile (64 rows x 32 K) staged through double-buffered LDS shared by the 4 waves;
// A fragments are wave-private and software-pipelined one stage ahead in registers,
// so no WMMA waits on a load issued in its own stage.

union Frag { u16x8 h[2]; v16bf v; };

__global__ __launch_bounds__(128) void k_gemm_bf16(
    const unsigned short* __restrict__ A, int lda,
    const unsigned short* __restrict__ Bw, int ldb,
    float* __restrict__ C, int ldc, int K,
    const float* __restrict__ bias,
    const float* __restrict__ rowScale, int rss,
    int act, int accum) {
  __shared__ unsigned short lb[2][64 * LSTRIDE];

  int tid  = threadIdx.x;
  int wave = tid >> 5;
  int lane = tid & 31;
  int l16  = lane & 15;
  int hi   = lane >> 4;
  int tileM = blockIdx.y * 64 + wave * 16;
  int tileN = blockIdx.x * 64;

  v8f acc[4] = {};

  // cooperative B-tile copy mapping: 128 threads x 32B = 64 rows x 32 bf16
  int crow = tid >> 1;                 // 0..63
  int ccol = (tid & 1) * 16;           // elem offset 0 or 16
  const unsigned short* bsrc = Bw + (long)(tileN + crow) * ldb + ccol;
  unsigned short* bdst0 = &lb[0][crow * LSTRIDE + ccol];
  unsigned short* bdst1 = &lb[1][crow * LSTRIDE + ccol];

  // A fragment source: lane<16 holds K {0..7,16..23}; lane>=16 holds K {8..15,24..31}
  const unsigned short* ap = A + (long)(tileM + l16) * lda + hi * 8;

  // prologue: stage-0 B into LDS, stage-0 A into registers
  u16x8 breg0 = *(const u16x8*)(bsrc);
  u16x8 breg1 = *(const u16x8*)(bsrc + 8);
  Frag a_cur;
  a_cur.h[0] = *(const u16x8*)(ap);
  a_cur.h[1] = *(const u16x8*)(ap + 16);
  *(u16x8*)(bdst0)     = breg0;
  *(u16x8*)(bdst0 + 8) = breg1;
  __syncthreads();

  int nstage = K >> 5;
  for (int s = 0; s < nstage; ++s) {
    int kb = s << 5;
    const unsigned short* lbase = (s & 1) ? lb[1] : lb[0];
    unsigned short* ndst = (s & 1) ? bdst0 : bdst1;
    bool more = (s + 1 < nstage);

    // issue next-stage global loads early (A and B overlap with WMMA below)
    Frag a_nxt;
    if (more) {
      __builtin_prefetch((const void*)(bsrc + kb + 96), 0, 0);  // stream weights into GL2
      breg0 = *(const u16x8*)(bsrc + kb + 32);
      breg1 = *(const u16x8*)(bsrc + kb + 40);
      a_nxt.h[0] = *(const u16x8*)(ap + kb + 32);
      a_nxt.h[1] = *(const u16x8*)(ap + kb + 48);
    }

#pragma unroll
    for (int nt = 0; nt < 4; ++nt) {
      const unsigned short* lp = lbase + (nt * 16 + l16) * LSTRIDE + hi * 8;
      Frag b;
      b.h[0] = *(const u16x8*)(lp);
      b.h[1] = *(const u16x8*)(lp + 16);
      acc[nt] = __builtin_amdgcn_wmma_f32_16x16x32_bf16(
          false, a_cur.v, false, b.v, (short)0, acc[nt], false, false);
    }

    if (more) {
      __syncthreads();                 // everyone done reading the buffer we overwrite
      *(u16x8*)(ndst)     = breg0;
      *(u16x8*)(ndst + 8) = breg1;
      __syncthreads();                 // stores visible before next stage reads
      a_cur = a_nxt;
    }
  }

  // C layout: VGPR j -> row tileM + hi*8 + j, col = tileN + nt*16 + l16
#pragma unroll
  for (int nt = 0; nt < 4; ++nt) {
    int col = tileN + nt * 16 + l16;
    float bs = bias ? bias[col] : 0.f;
#pragma unroll
    for (int j = 0; j < 8; ++j) {
      int row = tileM + hi * 8 + j;
      float v = acc[nt][j] + bs;
      if (act == 1)      v = silu_(v);
      else if (act == 2) v = (v > 20.f) ? v : log1pf(__expf(v));  // softplus
      if (rowScale) v *= rowScale[(long)row * rss];
      long idx = (long)row * ldc + col;
      if (accum) C[idx] += v; else C[idx] = v;
    }
  }
}

// ---------------- host-side helpers ----------------

static void gemm(hipStream_t s, const unsigned short* A, int lda,
                 const unsigned short* Bw, int ldb, float* C, int ldc,
                 int M, int N, int K, const float* bias,
                 const float* rowScale, int rss, int act, int accum) {
  dim3 g(N / 64, M / 64);
  k_gemm_bf16<<<g, dim3(128), 0, s>>>(A, lda, Bw, ldb, C, ldc, K, bias, rowScale, rss, act, accum);
}

static void cvt(hipStream_t s, const float* src, unsigned short* dst, long n) {
  long blocks = (n + 255) / 256;
  if (blocks > 8192) blocks = 8192;
  k_f2bf<<<dim3((unsigned)blocks), dim3(256), 0, s>>>(src, dst, n);
}

extern "C" void kernel_launch(void* const* d_in, const int* in_sizes, int n_in,
                              void* d_out, int out_size, void* d_ws, size_t ws_size,
                              hipStream_t stream) {
  (void)in_sizes; (void)n_in; (void)out_size; (void)ws_size;
  const float* x          = (const float*)d_in[0];
  const float* spatial_w  = (const float*)d_in[1];
  const float* spatial_b  = (const float*)d_in[2];
  const float* temporal_w = (const float*)d_in[3];
  const float* temporal_b = (const float*)d_in[4];
  const float* cls_token  = (const float*)d_in[5];
  const float* norm1_w    = (const float*)d_in[6];
  const float* in_proj_w  = (const float*)d_in[7];
  const float* conv_w_f   = (const float*)d_in[8];
  const float* conv_b_f   = (const float*)d_in[9];
  const float* conv_w_b   = (const float*)d_in[10];
  const float* conv_b_b   = (const float*)d_in[11];
  const float* xproj_w_f  = (const float*)d_in[12];
  const float* xproj_w_b  = (const float*)d_in[13];
  const float* dtproj_w_f = (const float*)d_in[14];
  const float* dtproj_b_f = (const float*)d_in[15];
  const float* dtproj_w_b = (const float*)d_in[16];
  const float* dtproj_b_b = (const float*)d_in[17];
  const float* A_log_f    = (const float*)d_in[18];
  const float* A_log_b    = (const float*)d_in[19];
  const float* Dp_f       = (const float*)d_in[20];
  const float* Dp_b       = (const float*)d_in[21];
  const float* out_proj_w = (const float*)d_in[22];
  const float* norm2_w    = (const float*)d_in[23];
  const float* gate_w     = (const float*)d_in[24];
  const float* e_w1       = (const float*)d_in[25];
  const float* e_w2       = (const float*)d_in[26];
  const float* cls_w      = (const float*)d_in[27];
  const float* cls_b      = (const float*)d_in[28];
  float* out = (float*)d_out;

  // ---- workspace layout ----
  char* base = (char*)d_ws;
  size_t off = 0;
  auto alloc = [&](size_t bytes) -> char* {
    char* p = base + off;
    off = (off + bytes + 255) & ~(size_t)255;
    return p;
  };
  float* T              = (float*)alloc((size_t)MP_ * D_ * 4);
  float* gates          = (float*)alloc((size_t)MP_ * E_ * 4);
  unsigned short* wbuf  = (unsigned short*)alloc((size_t)D_ * D_ * PATCH_ * 2); // max weight (temporal)
  char* arena = base + off;

  // stage-1 aliases (dead after T is built)
  float* y                 = (float*)(arena);                              // 49,152,000 B
  unsigned short* patches  = (unsigned short*)(arena + 49152000);          // 24,903,680 B
  float* tok               = (float*)(arena + 49152000 + 24903680);        //  2,490,368 B
  // per-block aliases (reuse arena)
  unsigned short* hb   = (unsigned short*)(arena + 0);
  float* xz            = (float*)(arena + 1245184);
  float* xcf           = (float*)(arena + 11206656);
  float* xcb           = (float*)(arena + 16187392);
  unsigned short* xc16 = (unsigned short*)(arena + 21168128);
  float* proj          = (float*)(arena + 23658496);
  unsigned short* dtrb = (unsigned short*)(arena + 23969792);
  float* dt            = (float*)(arena + 24047616);
  float* ysf           = (float*)(arena + 29028352);
  float* ysb           = (float*)(arena + 34009088);
  unsigned short* ub   = (unsigned short*)(arena + 38989824);
  float* t1            = (float*)(arena + 41480192);
  unsigned short* t1b  = (unsigned short*)(arena + 51441664);

  // ---- stem: spatial silu, patch-conv (as im2col GEMM), build token matrix ----
  {
    long n = (long)B_ * D_ * L_;
    k_spatial<<<dim3((unsigned)((n + 255) / 256)), dim3(256), 0, stream>>>(x, spatial_w, spatial_b, y);
  }
  {
    long n = (long)MP_ * D_ * PATCH_;
    k_im2col<<<dim3((unsigned)((n + 255) / 256)), dim3(256), 0, stream>>>(y, patches);
  }
  cvt(stream, temporal_w, wbuf, (long)D_ * D_ * PATCH_);
  gemm(stream, patches, D_ * PATCH_, wbuf, D_ * PATCH_, tok, D_,
       MP_, D_, D_ * PATCH_, temporal_b, nullptr, 0, 0, 0);
  {
    long n = (long)MP_ * D_;
    k_build_T<<<dim3((unsigned)((n + 255) / 256)), dim3(256), 0, stream>>>(tok, cls_token, T);
  }

  // ---- blocks ----
  for (int i = 0; i < NB_; ++i) {
    k_rms_bf16<<<dim3(MP_), dim3(256), 0, stream>>>(T, norm1_w + i * D_, hb);
    cvt(stream, in_proj_w + (size_t)i * 2 * DI_ * D_, wbuf, (long)2 * DI_ * D_);
    gemm(stream, hb, D_, wbuf, D_, xz, 2 * DI_, MP_, 2 * DI_, D_, nullptr, nullptr, 0, 0, 0);

    for (int dir = 0; dir < 2; ++dir) {
      const float* cw  = dir ? conv_w_b   : conv_w_f;
      const float* cb  = dir ? conv_b_b   : conv_b_f;
      const float* xw  = dir ? xproj_w_b  : xproj_w_f;
      const float* dw  = dir ? dtproj_w_b : dtproj_w_f;
      const float* db  = dir ? dtproj_b_b : dtproj_b_f;
      const float* Al  = dir ? A_log_b    : A_log_f;
      const float* Dpv = dir ? Dp_b       : Dp_f;
      float* xc = dir ? xcb : xcf;
      float* ys = dir ? ysb : ysf;
      {
        long n = (long)MP_ * DI_;
        k_dwconv_silu<<<dim3((unsigned)((n + 255) / 256)), dim3(256), 0, stream>>>(
            xz, cw + (size_t)i * DI_ * KC_, cb + i * DI_, xc, dir);
      }
      cvt(stream, xc, xc16, (long)MP_ * DI_);
      cvt(stream, xw + (size_t)i * (DTR_ + 2 * DS_) * DI_, wbuf, (long)(DTR_ + 2 * DS_) * DI_);
      gemm(stream, xc16, DI_, wbuf, DI_, proj, 64, MP_, 64, DI_, nullptr, nullptr, 0, 0, 0);
      {
        long n = (long)MP_ * DTR_;
        k_extract_dtr<<<dim3((unsigned)((n + 255) / 256)), dim3(256), 0, stream>>>(proj, dtrb);
      }
      cvt(stream, dw + (size_t)i * DI_ * DTR_, wbuf, (long)DI_ * DTR_);
      gemm(stream, dtrb, DTR_, wbuf, DTR_, dt, DI_, MP_, DI_, DTR_,
           db + i * DI_, nullptr, 0, 2 /*softplus*/, 0);
      k_ssm<<<dim3((B_ * DI_ + 255) / 256), dim3(256), 0, stream>>>(
          xc, dt, proj, Al + (size_t)i * DI_ * DS_, Dpv + i * DI_, ys);
    }

    {
      long n = (long)MP_ * DI_;
      k_combine<<<dim3((unsigned)((n + 255) / 256)), dim3(256), 0, stream>>>(ysf, ysb, xz, ub);
    }
    cvt(stream, out_proj_w + (size_t)i * D_ * DI_, wbuf, (long)D_ * DI_);
    gemm(stream, ub, DI_, wbuf, DI_, T, D_, MP_, D_, DI_, nullptr, nullptr, 0, 0, 1 /*T +=*/);

    // MoE
    k_rms_bf16<<<dim3(MP_), dim3(256), 0, stream>>>(T, norm2_w + i * D_, hb);
    k_gate<<<dim3((MP_ + 63) / 64), dim3(64), 0, stream>>>(hb, gate_w + (size_t)i * E_ * D_, gates);
    for (int e = 0; e < E_; ++e) {
      cvt(stream, e_w1 + ((size_t)i * E_ + e) * F_ * D_, wbuf, (long)F_ * D_);
      gemm(stream, hb, D_, wbuf, D_, t1, F_, MP_, F_, D_, nullptr, nullptr, 0, 1 /*silu*/, 0);
      cvt(stream, t1, t1b, (long)MP_ * F_);
      cvt(stream, e_w2 + ((size_t)i * E_ + e) * D_ * F_, wbuf, (long)D_ * F_);
      gemm(stream, t1b, F_, wbuf, F_, T, D_, MP_, D_, F_,
           nullptr, gates + e, E_, 0, 1 /*T += gate*expert*/);
    }
  }

  k_cls<<<dim3(1), dim3(64), 0, stream>>>(T, cls_w, cls_b, out);
}
